// Pair_attention_layer_29145648071158
// MI455X (gfx1250) — matrix operand order repaired and checked by reading, not compiled, not run on hardware
//
#include <hip/hip_runtime.h>

#define DEVINL __device__ __forceinline__

constexpr int N  = 2048;
constexpr int D  = 1024;
constexpr int H  = 16;
constexpr int DH = 64;

typedef __attribute__((ext_vector_type(16))) __bf16 v16bf;
typedef __attribute__((ext_vector_type(8)))  __bf16 v8bf;
typedef __attribute__((ext_vector_type(8)))  float  v8f;

DEVINL v8f wmma_bf16(v16bf a, v16bf b, v8f c) {
  return __builtin_amdgcn_wmma_f32_16x16x32_bf16(false, a, false, b, (short)0, c, false, false);
}

// Assemble a 16-element bf16 fragment from two contiguous 8-element (16B) chunks.
DEVINL v16bf ld_frag(const __bf16* p0, const __bf16* p1) {
  v8bf a = *(const v8bf*)p0;
  v8bf b = *(const v8bf*)p1;
  v16bf r;
#pragma unroll
  for (int i = 0; i < 8; ++i) { r[i] = a[i]; r[i + 8] = b[i]; }
  return r;
}
// Two 8-element chunks at p and p+16 — B-operand per-lane layout (matches A).
DEVINL v16bf ld_frag_c(const __bf16* p) { return ld_frag(p, p + 16); }

// ---------------------------------------------------------------- fp32 -> bf16
__global__ void cvt_kernel(const float* __restrict__ src, __bf16* __restrict__ dst, int n) {
  int i = blockIdx.x * blockDim.x + threadIdx.x;
  if (i < n) dst[i] = (__bf16)src[i];
}

// ---------------------------------------------------------------- QKV projections
// out[n, j] = sum_k X[n,k] * W[j,k] + bias[j].  One wave = one 32x32 output tile.
// Q, K stored row-major bf16 [N, D]; V stored transposed bf16 [D, N].
__global__ void __launch_bounds__(256) qkv_gemm_kernel(
    const __bf16* __restrict__ X,
    const __bf16* __restrict__ Wq, const __bf16* __restrict__ Wk, const __bf16* __restrict__ Wv,
    const float* __restrict__ bq, const float* __restrict__ bk, const float* __restrict__ bv,
    __bf16* __restrict__ Qo, __bf16* __restrict__ Ko, __bf16* __restrict__ Vt)
{
  const int lane  = threadIdx.x & 31;
  const int wid   = (blockIdx.x * blockDim.x + threadIdx.x) >> 5;
  const int tileM = wid & (N / 32 - 1);     // 64 tiles along M
  const int tileN = wid >> 6;               // 32 tiles along N
  const int m0 = tileM * 32, n0 = tileN * 32;
  const int mat = blockIdx.y;
  const __bf16* W    = (mat == 0) ? Wq : (mat == 1) ? Wk : Wv;
  const float*  bias = (mat == 0) ? bq : (mat == 1) ? bk : bv;

  const int akoff = (lane >= 16) ? 8 : 0;   // A-operand K-chunk select per lane half
  const int bcol  = lane & 15;              // B-operand column per lane
  const int bkoff = (lane >= 16) ? 8 : 0;   // B-operand K-chunk select

  const __bf16* a0p = X + (size_t)(m0 + (lane & 15)) * D + akoff;
  const __bf16* a1p = a0p + (size_t)16 * D;
  const __bf16* b0p = W + (size_t)(n0 + bcol) * D + bkoff;
  const __bf16* b1p = b0p + (size_t)16 * D;

  v8f acc00 = {}, acc01 = {}, acc10 = {}, acc11 = {};
  for (int k = 0; k < D; k += 32) {
    v16bf a0 = ld_frag(a0p + k, a0p + k + 16);
    v16bf a1 = ld_frag(a1p + k, a1p + k + 16);
    v16bf b0 = ld_frag_c(b0p + k);
    v16bf b1 = ld_frag_c(b1p + k);
    acc00 = wmma_bf16(a0, b0, acc00);
    acc01 = wmma_bf16(a0, b1, acc01);
    acc10 = wmma_bf16(a1, b0, acc10);
    acc11 = wmma_bf16(a1, b1, acc11);
  }

  const int   rbase = (lane >= 16) ? 8 : 0;
  const int   colLo = n0 + bcol, colHi = colLo + 16;
  const float bLo = bias[colLo], bHi = bias[colHi];

  if (mat < 2) {
    __bf16* O = (mat == 0) ? Qo : Ko;
#pragma unroll
    for (int r = 0; r < 8; ++r) {
      int row0 = m0 + rbase + r, row1 = row0 + 16;
      O[(size_t)row0 * D + colLo] = (__bf16)(acc00[r] + bLo);
      O[(size_t)row0 * D + colHi] = (__bf16)(acc01[r] + bHi);
      O[(size_t)row1 * D + colLo] = (__bf16)(acc10[r] + bLo);
      O[(size_t)row1 * D + colHi] = (__bf16)(acc11[r] + bHi);
    }
  } else {
    // V transposed: Vt[col, row]; each lane's 8 slots are consecutive rows -> packed 16B store.
    v8bf p00, p01, p10, p11;
#pragma unroll
    for (int r = 0; r < 8; ++r) {
      p00[r] = (__bf16)(acc00[r] + bLo);
      p01[r] = (__bf16)(acc01[r] + bHi);
      p10[r] = (__bf16)(acc10[r] + bLo);
      p11[r] = (__bf16)(acc11[r] + bHi);
    }
    *(v8bf*)(Vt + (size_t)colLo * N + m0 + rbase)      = p00;
    *(v8bf*)(Vt + (size_t)colHi * N + m0 + rbase)      = p01;
    *(v8bf*)(Vt + (size_t)colLo * N + m0 + 16 + rbase) = p10;
    *(v8bf*)(Vt + (size_t)colHi * N + m0 + 16 + rbase) = p11;
  }
}

// ---------------------------------------------------------------- flash attention
// One wave handles (head h, 16 query rows): streams 32 keys/iteration with
// online softmax; ctx accumulated in f32, written bf16 row-major [N, D].
__global__ void __launch_bounds__(256) attn_kernel(
    const __bf16* __restrict__ Q, const __bf16* __restrict__ Km,
    const __bf16* __restrict__ Vt, const float* __restrict__ mask,
    __bf16* __restrict__ Cx)
{
  __shared__ __bf16 plds[8][16 * 32];       // per-wave P staging (1 KB each)

  const int lane = threadIdx.x & 31;
  const int wloc = threadIdx.x >> 5;
  const int gw   = blockIdx.x * 8 + wloc;
  const int head = gw >> 7;                 // N/16 = 128 q-blocks per head
  const int m0   = (gw & 127) * 16;
  const int hoff = head * DH;

  const int akoff = (lane >= 16) ? 8 : 0;
  const int bcol  = lane & 15;
  const int bkoff = (lane >= 16) ? 8 : 0;

  // Q fragments for d=0..31 and d=32..63 (loaded once)
  const __bf16* qp = Q + (size_t)(m0 + (lane & 15)) * D + hoff;
  v16bf qa0 = ld_frag(qp + akoff,      qp + 16 + akoff);
  v16bf qa1 = ld_frag(qp + 32 + akoff, qp + 48 + akoff);

  v8f acc0 = {}, acc1 = {}, acc2 = {}, acc3 = {};
  float mrun[8], lrun[8];
#pragma unroll
  for (int r = 0; r < 8; ++r) { mrun[r] = -1e30f; lrun[r] = 0.0f; }

  __bf16* pb = plds[wloc];
  const int prow = (lane >= 16) ? 8 : 0;

  for (int kb = 0; kb < N; kb += 32) {
    // scores S[16 x 32] = Q (16x64) x K_blkT (64x32), scale 1/sqrt(64)
    const __bf16* kr0 = Km + (size_t)(kb + bcol) * D + hoff;
    const __bf16* kr1 = kr0 + (size_t)16 * D;
    v8f s0 = {}, s1 = {};
    s0 = wmma_bf16(qa0, ld_frag_c(kr0 + bkoff), s0);
    s0 = wmma_bf16(qa1, ld_frag_c(kr0 + 32 + bkoff), s0);
    s1 = wmma_bf16(qa0, ld_frag_c(kr1 + bkoff), s1);
    s1 = wmma_bf16(qa1, ld_frag_c(kr1 + 32 + bkoff), s1);

    const float madd0 = (1.0f - mask[kb + bcol]) * -10000.0f;
    const float madd1 = (1.0f - mask[kb + 16 + bcol]) * -10000.0f;

#pragma unroll
    for (int r = 0; r < 8; ++r) {
      float x0 = s0[r] * 0.125f + madd0;
      float x1 = s1[r] * 0.125f + madd1;
      // row max across the 16-lane half (xor masks <= 8 stay within the half)
      float mr = fmaxf(x0, x1);
      mr = fmaxf(mr, __shfl_xor(mr, 1));
      mr = fmaxf(mr, __shfl_xor(mr, 2));
      mr = fmaxf(mr, __shfl_xor(mr, 4));
      mr = fmaxf(mr, __shfl_xor(mr, 8));
      float mnew = fmaxf(mrun[r], mr);
      float corr = __expf(mrun[r] - mnew);
      mrun[r] = mnew;
      float p0 = __expf(x0 - mnew);
      float p1 = __expf(x1 - mnew);
      float rs = p0 + p1;
      rs += __shfl_xor(rs, 1);
      rs += __shfl_xor(rs, 2);
      rs += __shfl_xor(rs, 4);
      rs += __shfl_xor(rs, 8);
      lrun[r] = lrun[r] * corr + rs;
      acc0[r] *= corr; acc1[r] *= corr; acc2[r] *= corr; acc3[r] *= corr;
      s0[r] = p0; s1[r] = p1;
    }

    // C-layout -> A-layout transpose of P through LDS (wave-local, in-order DS)
#pragma unroll
    for (int r = 0; r < 8; ++r) {
      pb[(prow + r) * 32 + bcol]      = (__bf16)s0[r];
      pb[(prow + r) * 32 + 16 + bcol] = (__bf16)s1[r];
    }
    asm volatile("s_wait_dscnt 0x0" ::: "memory");
    const __bf16* pr = pb + (lane & 15) * 32;
    v16bf pa = ld_frag(pr + akoff, pr + 16 + akoff);

    // ctx += P (16x32) x V_blk (32x64); V fragments contiguous via Vt
    const __bf16* vr = Vt + (size_t)(hoff + bcol) * N + kb + bkoff;
    acc0 = wmma_bf16(pa, ld_frag_c(vr),                    acc0);
    acc1 = wmma_bf16(pa, ld_frag_c(vr + (size_t)16 * N),   acc1);
    acc2 = wmma_bf16(pa, ld_frag_c(vr + (size_t)32 * N),   acc2);
    acc3 = wmma_bf16(pa, ld_frag_c(vr + (size_t)48 * N),   acc3);
  }

  const int rbase = (lane >= 16) ? 8 : 0;
#pragma unroll
  for (int r = 0; r < 8; ++r) {
    float inv = 1.0f / lrun[r];
    __bf16* o = Cx + (size_t)(m0 + rbase + r) * D + hoff + bcol;
    o[0]  = (__bf16)(acc0[r] * inv);
    o[16] = (__bf16)(acc1[r] * inv);
    o[32] = (__bf16)(acc2[r] * inv);
    o[48] = (__bf16)(acc3[r] * inv);
  }
}

// ---------------------------------------------------------------- dense + tanh + residual
__global__ void __launch_bounds__(256) dense_kernel(
    const __bf16* __restrict__ Cx, const __bf16* __restrict__ Wd,
    const float* __restrict__ bd, const float* __restrict__ ht,
    float* __restrict__ sht)
{
  const int lane  = threadIdx.x & 31;
  const int wid   = (blockIdx.x * blockDim.x + threadIdx.x) >> 5;
  const int m0 = (wid & (N / 32 - 1)) * 32;
  const int n0 = (wid >> 6) * 32;

  const int akoff = (lane >= 16) ? 8 : 0;
  const int bcol  = lane & 15;
  const int bkoff = (lane >= 16) ? 8 : 0;

  const __bf16* a0p = Cx + (size_t)(m0 + (lane & 15)) * D + akoff;
  const __bf16* a1p = a0p + (size_t)16 * D;
  const __bf16* b0p = Wd + (size_t)(n0 + bcol) * D + bkoff;
  const __bf16* b1p = b0p + (size_t)16 * D;

  v8f acc00 = {}, acc01 = {}, acc10 = {}, acc11 = {};
  for (int k = 0; k < D; k += 32) {
    v16bf a0 = ld_frag(a0p + k, a0p + k + 16);
    v16bf a1 = ld_frag(a1p + k, a1p + k + 16);
    v16bf b0 = ld_frag_c(b0p + k);
    v16bf b1 = ld_frag_c(b1p + k);
    acc00 = wmma_bf16(a0, b0, acc00);
    acc01 = wmma_bf16(a0, b1, acc01);
    acc10 = wmma_bf16(a1, b0, acc10);
    acc11 = wmma_bf16(a1, b1, acc11);
  }

  const int   rbase = (lane >= 16) ? 8 : 0;
  const int   colLo = n0 + bcol, colHi = colLo + 16;
  const float bLo = bd[colLo], bHi = bd[colHi];
#pragma unroll
  for (int r = 0; r < 8; ++r) {
    int row0 = m0 + rbase + r, row1 = row0 + 16;
    sht[(size_t)row0 * D + colLo] = tanhf(acc00[r] + bLo) + ht[(size_t)row0 * D + colLo];
    sht[(size_t)row0 * D + colHi] = tanhf(acc01[r] + bHi) + ht[(size_t)row0 * D + colHi];
    sht[(size_t)row1 * D + colLo] = tanhf(acc10[r] + bLo) + ht[(size_t)row1 * D + colLo];
    sht[(size_t)row1 * D + colHi] = tanhf(acc11[r] + bHi) + ht[(size_t)row1 * D + colHi];
  }
}

// ---------------------------------------------------------------- row LayerNorm
__global__ void __launch_bounds__(256) ln_kernel(
    const float* __restrict__ sht, const float* __restrict__ gamma,
    const float* __restrict__ beta, float* __restrict__ out)
{
  __shared__ float rs[256], rq[256];
  const int row = blockIdx.x, t = threadIdx.x;
  const float* x = sht + (size_t)row * D;
  float v[4], s = 0.0f, s2 = 0.0f;
#pragma unroll
  for (int i = 0; i < 4; ++i) { v[i] = x[t + i * 256]; s += v[i]; s2 += v[i] * v[i]; }
  rs[t] = s; rq[t] = s2;
  __syncthreads();
  for (int off = 128; off > 0; off >>= 1) {
    if (t < off) { rs[t] += rs[t + off]; rq[t] += rq[t + off]; }
    __syncthreads();
  }
  const float mu  = rs[0] * (1.0f / D);
  const float var = rq[0] * (1.0f / D) - mu * mu;
  const float rn  = rsqrtf(var + 1e-5f);
  float* o = out + (size_t)row * D;
#pragma unroll
  for (int i = 0; i < 4; ++i) {
    int c = t + i * 256;
    o[c] = (v[i] - mu) * rn * gamma[c] + beta[c];
  }
}

// ----------------------------------------------------------------------------
extern "C" void kernel_launch(void* const* d_in, const int* in_sizes, int n_in,
                              void* d_out, int out_size, void* d_ws, size_t ws_size,
                              hipStream_t stream) {
  const float* ht    = (const float*)d_in[0];
  const float* mask  = (const float*)d_in[1];
  const float* Wq    = (const float*)d_in[2];
  const float* bq    = (const float*)d_in[3];
  const float* Wk    = (const float*)d_in[4];
  const float* bk    = (const float*)d_in[5];
  const float* Wv    = (const float*)d_in[6];
  const float* bv    = (const float*)d_in[7];
  const float* Wd    = (const float*)d_in[8];
  const float* bd    = (const float*)d_in[9];
  const float* gamma = (const float*)d_in[10];
  const float* beta  = (const float*)d_in[11];

  __bf16* hb  = (__bf16*)d_ws;
  __bf16* wqb = hb  + (size_t)N * D;
  __bf16* wkb = wqb + (size_t)D * D;
  __bf16* wvb = wkb + (size_t)D * D;
  __bf16* wdb = wvb + (size_t)D * D;
  __bf16* Qb  = wdb + (size_t)D * D;
  __bf16* Kb  = Qb  + (size_t)N * D;
  __bf16* Vt  = Kb  + (size_t)N * D;
  __bf16* Cx  = Vt  + (size_t)N * D;
  float*  sht = (float*)(Cx + (size_t)N * D);

  cvt_kernel<<<(N * D + 255) / 256, 256, 0, stream>>>(ht, hb, N * D);
  cvt_kernel<<<(D * D + 255) / 256, 256, 0, stream>>>(Wq, wqb, D * D);
  cvt_kernel<<<(D * D + 255) / 256, 256, 0, stream>>>(Wk, wkb, D * D);
  cvt_kernel<<<(D * D + 255) / 256, 256, 0, stream>>>(Wv, wvb, D * D);
  cvt_kernel<<<(D * D + 255) / 256, 256, 0, stream>>>(Wd, wdb, D * D);

  // 2048 tiles / 8 waves per block = 256 blocks; y selects Q/K/V
  qkv_gemm_kernel<<<dim3(256, 3), 256, 0, stream>>>(hb, wqb, wkb, wvb, bq, bk, bv, Qb, Kb, Vt);
  // 16 heads * 128 query-blocks = 2048 waves
  attn_kernel<<<256, 256, 0, stream>>>(Qb, Kb, Vt, mask, Cx);
  dense_kernel<<<256, 256, 0, stream>>>(Cx, wdb, bd, ht, sht);
  ln_kernel<<<N, 256, 0, stream>>>(sht, gamma, beta, (float*)d_out);
}
